// RoPEMHA_10565619548314
// MI455X (gfx1250) — matrix operand order repaired and checked by reading, not compiled, not run on hardware
//
#include <hip/hip_runtime.h>
#include <hip/hip_bf16.h>

// ---------------- problem constants ----------------
#define Bq   2
#define Nq   2048
#define Cq   768
#define Hq   12
#define Dq   64
#define Mq   (Bq*Nq)          // 4096 rows
#define QKVN (3*Cq)           // 2304

typedef __attribute__((ext_vector_type(16))) _Float16 v16h;
typedef __attribute__((ext_vector_type(8)))  _Float16 h8;
typedef __attribute__((ext_vector_type(8)))  float    v8f;

// ---------------- WMMA helpers ----------------
__device__ __forceinline__ v8f wmma_f16(v16h a, v16h b, v8f c) {
  // D = A(16x32 f16) * B(32x16 f16) + C(16x16 f32)
  return __builtin_amdgcn_wmma_f32_16x16x32_f16(false, a, false, b, (short)0, c,
                                                false, false);
}

// A-fragment (16x32, MxK): lane holds row M=lane%16.
// half h: K = 16*(lane/16) + h -> one contiguous 32B run.
__device__ __forceinline__ v16h load_frag_a(const _Float16* p) {
  h8 lo = *(const h8*)p;
  h8 hi = *(const h8*)(p + 8);
  v16h r;
#pragma unroll
  for (int i = 0; i < 8; ++i) { r[i] = lo[i]; r[i + 8] = hi[i]; }
  return r;
}

// B-fragment (32x16, KxN), B stored [N][K]: lane holds col N=lane%16,
// half h: K = 16*(lane/16) + h -> one contiguous 32B run.
__device__ __forceinline__ v16h load_frag_b(const _Float16* p) {
  h8 lo = *(const h8*)p;
  h8 hi = *(const h8*)(p + 8);
  v16h r;
#pragma unroll
  for (int i = 0; i < 8; ++i) { r[i] = lo[i]; r[i + 8] = hi[i]; }
  return r;
}

// ---------------- gfx1250 async copy: global -> LDS (ASYNCcnt) ------------
// ISA 15.18.3 GLOBAL_LOAD_ASYNC_TO_LDS_B128: vdst = per-lane LDS byte address,
// vaddr = 64-bit global address. Generic shared pointers carry the LDS offset
// in their low 32 bits (ISA 10.2).
__device__ __forceinline__ void async_copy_b128(void* lds, const _Float16* g) {
  uint32_t lo = (uint32_t)(uintptr_t)lds;
  uint64_t ga = (uint64_t)(uintptr_t)g;
  asm volatile("global_load_async_to_lds_b128 %0, %1, off"
               :: "v"(lo), "v"(ga) : "memory");
}
__device__ __forceinline__ void wait_async0() {
#if __has_builtin(__builtin_amdgcn_s_wait_asynccnt)
  __builtin_amdgcn_s_wait_asynccnt(0);
#else
  asm volatile("s_wait_asynccnt 0x0" ::: "memory");
#endif
}

// ---------------- prep kernels ----------------
__global__ void cvt_f32_to_f16_kernel(const float* __restrict__ src,
                                      _Float16* __restrict__ dst, int n) {
  int i = blockIdx.x * blockDim.x + threadIdx.x;
  int stride = gridDim.x * blockDim.x;
  for (; i < n; i += stride) dst[i] = (_Float16)src[i];
}

// src[rows][cols] f32 -> dst[cols][rows] f16 (LDS-tiled transpose)
__global__ void transpose_f32_to_f16_kernel(const float* __restrict__ src,
                                            _Float16* __restrict__ dst,
                                            int rows, int cols) {
  __shared__ float tile[32][33];
  const int c0 = blockIdx.x * 32, r0 = blockIdx.y * 32;
  const int tx = threadIdx.x & 31, ty = threadIdx.x >> 5;
#pragma unroll
  for (int rr = ty; rr < 32; rr += 8) {
    int r = r0 + rr, c = c0 + tx;
    tile[rr][tx] = (r < rows && c < cols) ? src[r * cols + c] : 0.f;
  }
  __syncthreads();
#pragma unroll
  for (int cc = ty; cc < 32; cc += 8) {
    int c = c0 + cc, r = r0 + tx;
    if (c < cols && r < rows) dst[c * rows + r] = (_Float16)tile[tx][cc];
  }
}

// ---------------- QKV GEMM + bias + RoPE + scatter ----------------
// Wave tile 32(M) x 64(N): 8 WMMAs per K-step, B fragment reused by 2 WMMAs.
__global__ __launch_bounds__(256, 1)
void qkv_rope_kernel(const _Float16* __restrict__ xh,
                     const _Float16* __restrict__ wt,
                     const float* __restrict__ bqkv,
                     const float* __restrict__ sinp,
                     const float* __restrict__ cosp,
                     _Float16* __restrict__ qh,
                     _Float16* __restrict__ kh,
                     _Float16* __restrict__ vt) {
  const int lane = threadIdx.x & 31, wid = threadIdx.x >> 5;
  const int ll = lane & 15, lh = lane >> 4;
  const int m0 = blockIdx.x * 256 + wid * 32;   // wave's 32 rows
  const int n0 = blockIdx.y * 64;               // block's 64 cols (one head slice)

  v8f acc[2][4] = {};
  const _Float16* arow = xh + (size_t)(m0 + ll) * Cq + 16 * lh;

#pragma unroll 2
  for (int kk = 0; kk < Cq; kk += 32) {
    __builtin_prefetch(arow + kk + 128, 0, 1);        // global_prefetch_b8
    v16h a0 = load_frag_a(arow + kk);
    v16h a1 = load_frag_a(arow + 16 * Cq + kk);
#pragma unroll
    for (int t = 0; t < 4; ++t) {
      const _Float16* bp = wt + (size_t)(n0 + t * 16 + ll) * Cq + kk + 16 * lh;
      v16h b = load_frag_b(bp);
      acc[0][t] = wmma_f16(a0, b, acc[0][t]);
      acc[1][t] = wmma_f16(a1, b, acc[1][t]);
    }
  }

  // ---- epilogue: block tile lies entirely in one section & one head ----
  const int sec = n0 / Cq;            // 0=q 1=k 2=v
  const int h   = (n0 % Cq) / Dq;
  const int batch = m0 / Nq, tok0 = m0 % Nq;
  float bias[4];
#pragma unroll
  for (int t = 0; t < 4; ++t) bias[t] = bqkv[n0 + t * 16 + ll];

  if (sec < 2) {
    _Float16* dst = (sec == 0 ? qh : kh) + ((size_t)(batch * Hq + h) * Nq) * Dq;
    const float scale = (sec == 0) ? 0.125f : 1.0f;   // D^-1/2 folded into q
#pragma unroll
    for (int mi = 0; mi < 2; ++mi)
#pragma unroll
      for (int r = 0; r < 8; ++r) {
        const int tok = tok0 + mi * 16 + r + 8 * lh;
#pragma unroll
        for (int t = 0; t < 4; ++t) {
          const int d = t * 16 + ll;
          float val = acc[mi][t][r] + bias[t];
          float pv  = acc[mi][t ^ 2][r] + bias[t ^ 2];   // paired d +/- 32
          float sn = sinp[tok * Dq + d], cs = cosp[tok * Dq + d];
          float rot = (d < 32) ? -pv : pv;               // rotate_half
          dst[(size_t)tok * Dq + d] = (_Float16)((val * cs + rot * sn) * scale);
        }
      }
  } else {
    // v: store transposed [d][tok]; 8 consecutive tokens pack into one b128
    _Float16* dst = vt + ((size_t)(batch * Hq + h) * Dq) * Nq;
#pragma unroll
    for (int mi = 0; mi < 2; ++mi)
#pragma unroll
      for (int t = 0; t < 4; ++t) {
        const int d = t * 16 + ll;
        h8 pk;
#pragma unroll
        for (int r = 0; r < 8; ++r) pk[r] = (_Float16)(acc[mi][t][r] + bias[t]);
        *(h8*)(dst + (size_t)d * Nq + tok0 + mi * 16 + 8 * lh) = pk;
      }
  }
}

// ---------------- flash attention with async double-buffered K/V ----------
// 128 threads = 4 waves; wave owns 16 q rows x D=64 accumulator.
// K/V tiles staged once per block through LDS via global_load_async_to_lds.
__global__ __launch_bounds__(128, 1)
void attn_kernel(const _Float16* __restrict__ qh,
                 const _Float16* __restrict__ kh,
                 const _Float16* __restrict__ vt,
                 _Float16* __restrict__ yh) {
  const int lane = threadIdx.x & 31, wid = threadIdx.x >> 5;
  const int ll = lane & 15, lh = lane >> 4;
  const int b = blockIdx.z, h = blockIdx.y;
  const int q0 = blockIdx.x * 64 + wid * 16;

  const _Float16* qp = qh + ((size_t)(b * Hq + h) * Nq) * Dq;
  const _Float16* kp = kh + ((size_t)(b * Hq + h) * Nq) * Dq;
  const _Float16* vp = vt + ((size_t)(b * Hq + h) * Dq) * Nq;

  __shared__ __align__(16) _Float16 Kt[2][32][64];   // [tok][d]
  __shared__ __align__(16) _Float16 Vt[2][64][32];   // [d][tok]
  __shared__ __align__(16) _Float16 Plds[4][16][32]; // per-wave P restage

  // wave-cooperative async stage of one 32-token K/V tile (4 b128/lane)
  auto stage_tiles = [&](int buf, int t0) {
#pragma unroll
    for (int j = 0; j < 2; ++j) {               // K: 256 chunks of 16B
      int c = (wid * 2 + j) * 32 + lane;
      int row = c >> 3, q = c & 7;
      async_copy_b128(&Kt[buf][row][q * 8], kp + (size_t)(t0 + row) * Dq + q * 8);
    }
#pragma unroll
    for (int j = 0; j < 2; ++j) {               // V: 256 chunks of 16B
      int c = (wid * 2 + j) * 32 + lane;
      int row = c >> 2, q = c & 3;
      async_copy_b128(&Vt[buf][row][q * 8], vp + (size_t)row * Nq + t0 + q * 8);
    }
  };

  // loop-invariant q fragments: d=0..31 and d=32..63
  const v16h qf0 = load_frag_a(qp + (size_t)(q0 + ll) * Dq + 16 * lh);
  const v16h qf1 = load_frag_a(qp + (size_t)(q0 + ll) * Dq + 32 + 16 * lh);

  v8f acc[4] = {};
  float mr[8], lr[8];
#pragma unroll
  for (int r = 0; r < 8; ++r) { mr[r] = -3.0e38f; lr[r] = 0.f; }

  stage_tiles(0, 0);
  wait_async0();
  __syncthreads();

  const v8f zero = {};
  for (int t0 = 0; t0 < Nq; t0 += 32) {
    const int buf = (t0 >> 5) & 1;
    if (t0 + 32 < Nq) stage_tiles(buf ^ 1, t0 + 32);   // prefetch next tile

    // S = q k^T for 32 key tokens (two 16x16 tiles, K-dim = d in 2 steps)
    v8f s0 = wmma_f16(qf0, load_frag_b(&Kt[buf][ll][16 * lh]), zero);
    s0     = wmma_f16(qf1, load_frag_b(&Kt[buf][ll][32 + 16 * lh]), s0);
    v8f s1 = wmma_f16(qf0, load_frag_b(&Kt[buf][16 + ll][16 * lh]), zero);
    s1     = wmma_f16(qf1, load_frag_b(&Kt[buf][16 + ll][32 + 16 * lh]), s1);

    // online softmax: row = r + 8*(lane/16), spread over a 16-lane group
    float alpha[8];
#pragma unroll
    for (int r = 0; r < 8; ++r) {
      float x = fmaxf(s0[r], s1[r]);
      x = fmaxf(x, __shfl_xor(x, 1, 32));
      x = fmaxf(x, __shfl_xor(x, 2, 32));
      x = fmaxf(x, __shfl_xor(x, 4, 32));
      x = fmaxf(x, __shfl_xor(x, 8, 32));
      const float mnew = fmaxf(mr[r], x);
      const float p0 = __expf(s0[r] - mnew);
      const float p1 = __expf(s1[r] - mnew);
      float rs = p0 + p1;
      rs += __shfl_xor(rs, 1, 32);
      rs += __shfl_xor(rs, 2, 32);
      rs += __shfl_xor(rs, 4, 32);
      rs += __shfl_xor(rs, 8, 32);
      alpha[r] = __expf(mr[r] - mnew);
      lr[r] = lr[r] * alpha[r] + rs;
      mr[r] = mnew;
      Plds[wid][r + 8 * lh][ll]      = (_Float16)p0;   // restage P: C->A layout
      Plds[wid][r + 8 * lh][16 + ll] = (_Float16)p1;
    }
    __syncthreads();   // P visibility across the wave's lanes

#pragma unroll
    for (int t = 0; t < 4; ++t)
#pragma unroll
      for (int r = 0; r < 8; ++r) acc[t][r] *= alpha[r];

    const v16h pa = load_frag_a(&Plds[wid][ll][16 * lh]);
#pragma unroll
    for (int t = 0; t < 4; ++t)
      acc[t] = wmma_f16(pa, load_frag_b(&Vt[buf][t * 16 + ll][16 * lh]), acc[t]);

    wait_async0();     // own async copies for next tile done
    __syncthreads();   // whole block done -> safe to flip buffers
  }

  // y = acc / l  -> yh [B][N][C] f16 (head slice h*64)
#pragma unroll
  for (int r = 0; r < 8; ++r) {
    const int tok = q0 + r + 8 * lh;
    const float inv = 1.0f / lr[r];
#pragma unroll
    for (int t = 0; t < 4; ++t)
      yh[((size_t)(b * Nq + tok)) * Cq + h * Dq + t * 16 + ll] =
          (_Float16)(acc[t][r] * inv);
  }
}

// ---------------- output projection ----------------
__global__ __launch_bounds__(256, 1)
void out_gemm_kernel(const _Float16* __restrict__ yh,
                     const _Float16* __restrict__ wt,   // [768][768] = W_out^T
                     const float* __restrict__ bout,
                     float* __restrict__ out) {
  const int lane = threadIdx.x & 31, wid = threadIdx.x >> 5;
  const int ll = lane & 15, lh = lane >> 4;
  const int m0 = blockIdx.x * 256 + wid * 32;
  const int n0 = blockIdx.y * 64;

  v8f acc[2][4] = {};
  const _Float16* arow = yh + (size_t)(m0 + ll) * Cq + 16 * lh;
#pragma unroll 2
  for (int kk = 0; kk < Cq; kk += 32) {
    __builtin_prefetch(arow + kk + 128, 0, 1);
    v16h a0 = load_frag_a(arow + kk);
    v16h a1 = load_frag_a(arow + 16 * Cq + kk);
#pragma unroll
    for (int t = 0; t < 4; ++t) {
      const _Float16* bp = wt + (size_t)(n0 + t * 16 + ll) * Cq + kk + 16 * lh;
      v16h b = load_frag_b(bp);
      acc[0][t] = wmma_f16(a0, b, acc[0][t]);
      acc[1][t] = wmma_f16(a1, b, acc[1][t]);
    }
  }
  float bias[4];
#pragma unroll
  for (int t = 0; t < 4; ++t) bias[t] = bout[n0 + t * 16 + ll];
#pragma unroll
  for (int mi = 0; mi < 2; ++mi)
#pragma unroll
    for (int r = 0; r < 8; ++r) {
      const int row = m0 + mi * 16 + r + 8 * lh;
#pragma unroll
      for (int t = 0; t < 4; ++t)
        out[(size_t)row * Cq + n0 + t * 16 + ll] = acc[mi][t][r] + bias[t];
    }
}

// ---------------- workspace layout (f16 elements) ----------------
static const size_t OFF_XH    = 0;                                  // 4096*768
static const size_t OFF_WQKVT = OFF_XH    + (size_t)Mq * Cq;        // 2304*768
static const size_t OFF_WOUTT = OFF_WQKVT + (size_t)QKVN * Cq;      // 768*768
static const size_t OFF_Q     = OFF_WOUTT + (size_t)Cq * Cq;        // B*H*N*D
static const size_t OFF_K     = OFF_Q     + (size_t)Bq * Hq * Nq * Dq;
static const size_t OFF_VT    = OFF_K     + (size_t)Bq * Hq * Nq * Dq;
static const size_t OFF_YH    = OFF_VT    + (size_t)Bq * Hq * Nq * Dq;

extern "C" void kernel_launch(void* const* d_in, const int* in_sizes, int n_in,
                              void* d_out, int out_size, void* d_ws, size_t ws_size,
                              hipStream_t stream) {
  const float* x    = (const float*)d_in[0];
  const float* sinp = (const float*)d_in[1];
  const float* cosp = (const float*)d_in[2];
  const float* Wqkv = (const float*)d_in[3];
  const float* bqkv = (const float*)d_in[4];
  const float* Wout = (const float*)d_in[5];
  const float* bout = (const float*)d_in[6];
  float* out = (float*)d_out;

  _Float16* ws    = (_Float16*)d_ws;
  _Float16* xh    = ws + OFF_XH;
  _Float16* wqkvt = ws + OFF_WQKVT;
  _Float16* woutt = ws + OFF_WOUTT;
  _Float16* qh    = ws + OFF_Q;
  _Float16* kh    = ws + OFF_K;
  _Float16* vt    = ws + OFF_VT;
  _Float16* yh    = ws + OFF_YH;

  // 1) precision conversion / weight transposition
  cvt_f32_to_f16_kernel<<<2048, 256, 0, stream>>>(x, xh, Mq * Cq);
  transpose_f32_to_f16_kernel<<<dim3(QKVN / 32, Cq / 32), 256, 0, stream>>>(
      Wqkv, wqkvt, Cq, QKVN);
  transpose_f32_to_f16_kernel<<<dim3(Cq / 32, Cq / 32), 256, 0, stream>>>(
      Wout, woutt, Cq, Cq);

  // 2) fused QKV GEMM + bias + RoPE + q-scale + v-transpose
  qkv_rope_kernel<<<dim3(Mq / 256, QKVN / 64), 256, 0, stream>>>(
      xh, wqkvt, bqkv, sinp, cosp, qh, kh, vt);

  // 3) flash attention (async double-buffered K/V through LDS)
  attn_kernel<<<dim3(Nq / 64, Hq, Bq), 128, 0, stream>>>(qh, kh, vt, yh);

  // 4) output projection + bias
  out_gemm_kernel<<<dim3(Mq / 256, Cq / 64), 256, 0, stream>>>(yh, woutt, bout, out);
}
